// SelfAttentionWide_15504831939287
// MI455X (gfx1250) — matrix operand order repaired and checked by reading, not compiled, not run on hardware
//
#include <hip/hip_runtime.h>
#include <hip/hip_bf16.h>
#include <math.h>

// ---------------------------------------------------------------------------
// Types for WMMA fragments (gfx1250 wave32 layouts, see CDNA5 ISA)
// ---------------------------------------------------------------------------
typedef __bf16        bf16x16 __attribute__((ext_vector_type(16)));
typedef float         f32x8   __attribute__((ext_vector_type(8)));
typedef float         f32x4   __attribute__((ext_vector_type(4)));
typedef unsigned int  u32x4   __attribute__((ext_vector_type(4)));
typedef int           i32x4   __attribute__((ext_vector_type(4)));
typedef int           i32x8   __attribute__((ext_vector_type(8)));

struct U32x8 { u32x4 lo, hi; };

static __device__ __forceinline__ unsigned short f2bf(float f) {
    unsigned int u = __float_as_uint(f);
    u += 0x7fffu + ((u >> 16) & 1u);           // round-to-nearest-even
    return (unsigned short)(u >> 16);
}

static __device__ __forceinline__ bf16x16 make_frag(u32x4 lo, u32x4 hi) {
    U32x8 t{lo, hi};
    return __builtin_bit_cast(bf16x16, t);
}

// ---------------------------------------------------------------------------
// CDNA5 async global->LDS copy (ASYNCcnt-tracked, CDNA5 ISA)
// LDS address = low 32 bits of the generic pointer (aperture mapping, ISA 10.2)
// ---------------------------------------------------------------------------
static __device__ __forceinline__ void async_ld_b128(void* lds, const void* g) {
    unsigned l = (unsigned)(size_t)lds;
    asm volatile("global_load_async_to_lds_b128 %0, %1, off"
                 :: "v"(l), "v"(g) : "memory");
}
static __device__ __forceinline__ void wait_async0() {
    asm volatile("s_wait_asynccnt 0x0" ::: "memory");
}

// A operand (16x32 MxK), storage row-major [m][k] (LDS).
static __device__ __forceinline__ bf16x16
frag_a_lds(const unsigned short* s, int stride, int mBase, int kBase, int lane) {
    int half = lane >> 4, l = lane & 15;
    const unsigned short* row = s + (mBase + l) * stride;
    u32x4 p0 = *(const u32x4*)(row + kBase + half * 8);
    u32x4 p1 = *(const u32x4*)(row + kBase + 16 + half * 8);
    return make_frag(p0, p1);
}

// B operand (32x16 KxN), storage rows are the N dimension: [n][k] contiguous-k.
static __device__ __forceinline__ bf16x16
frag_b_lds(const unsigned short* s, int stride, int nBase, int kBase, int lane) {
    int half = lane >> 4, l = lane & 15;
    const unsigned short* row = s + (nBase + l) * stride;
    u32x4 p0 = *(const u32x4*)(row + kBase + half * 8);
    u32x4 p1 = *(const u32x4*)(row + kBase + 16 + half * 8);
    return make_frag(p0, p1);
}

// Same B layout reading straight from global bf16 [n][k] (per-lane 16B reads)
static __device__ __forceinline__ bf16x16
frag_b_glb(const unsigned short* g, size_t stride, int nBase, int kBase, int lane) {
    int half = lane >> 4, l = lane & 15;
    const unsigned short* row = g + (size_t)(nBase + l) * stride;
    u32x4 p0 = *(const u32x4*)(row + kBase + half * 8);
    u32x4 p1 = *(const u32x4*)(row + kBase + 16 + half * 8);
    return make_frag(p0, p1);
}

// ---------------------------------------------------------------------------
// fp32 -> bf16 bulk conversion (8 elems / thread)
// ---------------------------------------------------------------------------
__global__ __launch_bounds__(256)
void cvt_bf16(const float* __restrict__ src, unsigned short* __restrict__ dst, int n) {
    int i = (blockIdx.x * 256 + threadIdx.x) * 8;
    if (i + 8 > n) return;
    f32x4 a = *(const f32x4*)(src + i);
    f32x4 b = *(const f32x4*)(src + i + 4);
    u32x4 o;
    o[0] = (unsigned)f2bf(a[0]) | ((unsigned)f2bf(a[1]) << 16);
    o[1] = (unsigned)f2bf(a[2]) | ((unsigned)f2bf(a[3]) << 16);
    o[2] = (unsigned)f2bf(b[0]) | ((unsigned)f2bf(b[1]) << 16);
    o[3] = (unsigned)f2bf(b[2]) | ((unsigned)f2bf(b[3]) << 16);
    *(u32x4*)(dst + i) = o;
}

// ---------------------------------------------------------------------------
// GEMM: Out(M,N) = A(M,K) * W(N,K)^T, bf16 in, f32 WMMA accumulate.
// Async-to-LDS double-buffered staging. 256 thr = 8 waves (2x4), wave 64x32.
//   OUT_F32 : fp32 + bias store (unify) else bf16
//   V_TRANS : bf16 store transposed per-batch [b][n][t] (for V^T)
// ---------------------------------------------------------------------------
#define BM 128
#define BN 128
#define BK 32
#define GST 40   // LDS row stride (32+8) -> 80B, conflict-free & 16B aligned

template<bool OUT_F32, bool V_TRANS>
__global__ __launch_bounds__(256)
void gemm_wmma(const unsigned short* __restrict__ A,
               const unsigned short* __restrict__ W,
               void* __restrict__ Outv, const float* __restrict__ bias,
               float outScale, int M, int N, int K) {
    __shared__ unsigned short sA[2][BM * GST];
    __shared__ unsigned short sW[2][BN * GST];

    const int tid  = threadIdx.x;
    const int lane = tid & 31, w = tid >> 5;
    const int wm = w >> 2, wn = w & 3;                 // 2 x 4 wave grid
    const int bm = blockIdx.x * BM, bn = blockIdx.y * BN;

    const int lr = tid >> 1;                           // 0..127
    const int lc = (tid & 1) * 16;                     // 0 or 16

    f32x8 acc[4][2] = {};

    // stage tile k0 into LDS buffer `buf` (async, 4x b128 per thread)
    auto stage = [&](int buf, int k0) {
        const unsigned short* ga = A + (size_t)(bm + lr) * K + k0 + lc;
        const unsigned short* gw = W + (size_t)(bn + lr) * K + k0 + lc;
        unsigned short* la = &sA[buf][lr * GST + lc];
        unsigned short* lw = &sW[buf][lr * GST + lc];
        async_ld_b128(la,     ga);
        async_ld_b128(la + 8, ga + 8);
        async_ld_b128(lw,     gw);
        async_ld_b128(lw + 8, gw + 8);
    };

    stage(0, 0);
    wait_async0();
    __syncthreads();

    int buf = 0;
    for (int k0 = 0; k0 < K; k0 += BK) {
        if (k0 + BK < K) stage(buf ^ 1, k0 + BK);      // overlap next fetch

        bf16x16 af[4], bfr[2];
#pragma unroll
        for (int mi = 0; mi < 4; ++mi)
            af[mi] = frag_a_lds(sA[buf], GST, wm * 64 + mi * 16, 0, lane);
#pragma unroll
        for (int ni = 0; ni < 2; ++ni)
            bfr[ni] = frag_b_lds(sW[buf], GST, wn * 32 + ni * 16, 0, lane);
#pragma unroll
        for (int mi = 0; mi < 4; ++mi)
#pragma unroll
            for (int ni = 0; ni < 2; ++ni)
                acc[mi][ni] = __builtin_amdgcn_wmma_f32_16x16x32_bf16(
                    false, af[mi], false, bfr[ni], (short)0, acc[mi][ni],
                    false, false);

        wait_async0();
        __syncthreads();
        buf ^= 1;
    }

    // store (C tile: lane<16 -> M=j, lane>=16 -> M=j+8; N = lane&15)
    const int half = lane >> 4, cl = lane & 15;
#pragma unroll
    for (int mi = 0; mi < 4; ++mi)
#pragma unroll
        for (int ni = 0; ni < 2; ++ni)
#pragma unroll
            for (int j = 0; j < 8; ++j) {
                int m = bm + wm * 64 + mi * 16 + j + half * 8;
                int n = bn + wn * 32 + ni * 16 + cl;
                float v = acc[mi][ni][j] * outScale;
                if (OUT_F32) {
                    ((float*)Outv)[(size_t)m * N + n] = v + (bias ? bias[n] : 0.0f);
                } else if (V_TRANS) {
                    ((unsigned short*)Outv)
                        [((size_t)(m >> 10) * 4096 + n) * 1024 + (m & 1023)] = f2bf(v);
                } else {
                    ((unsigned short*)Outv)[(size_t)m * N + n] = f2bf(v);
                }
            }
}

// ---------------------------------------------------------------------------
// Flash attention per (b,h): Q,K bf16 [b][t][h][e]; Vt bf16 [b][h][e][t].
// 256 threads (8 waves), 32 query rows/block, KV tiles of 64, online softmax.
// Q tile staged with the Tensor Data Mover (TENSORcnt-tracked DMA).
// ---------------------------------------------------------------------------
#define FA_M   32
#define FA_KV  64
#define SQ_ST  520   // 512 + 8  (matches TDM pad: 1024B rows + 16B pad)
#define SS_ST  68    // 64 + 4 (floats)
#define SP_ST  72    // 64 + 8

__global__ __launch_bounds__(256)
void flash_attn(const unsigned short* __restrict__ Q,
                const unsigned short* __restrict__ Km,
                const unsigned short* __restrict__ Vt,
                unsigned short* __restrict__ O) {
    __shared__ unsigned short sQ[FA_M * SQ_ST];
    __shared__ float          sS[FA_M * SS_ST];
    __shared__ unsigned short sP[FA_M * SP_ST];
    __shared__ float          sM[FA_M], sL[FA_M], sF[FA_M];

    const int tid = threadIdx.x, lane = tid & 31, w = tid >> 5;
    const int qt = blockIdx.x, h = blockIdx.y, b = blockIdx.z;
    const int qBase = qt * FA_M;

    const unsigned short* qB = Q  + ((size_t)b * 1024 * 8 + h) * 512; // + t*4096 + e
    const unsigned short* kB = Km + ((size_t)b * 1024 * 8 + h) * 512;
    const unsigned short* vB = Vt + (size_t)(b * 8 + h) * 512 * 1024; // + e*1024 + t

    // ---- stage Q tile (32 x 512 bf16) via TDM: one D# issued by wave 0 ----
    // D# per CDNA5 ISA: 2D tile, data_size=2B,
    // tensor row stride 4096 elems, LDS pad 4 DWORDs every 256 DWORDs
    // (= +16B per 1024B row -> LDS row stride 520 elements == SQ_ST).
    if (tid < 32) {
        size_t gaddr = (size_t)(qB + (size_t)qBase * 4096);
        u32x4 g0;
        g0[0] = 1u;                                    // count=1, user mode
        g0[1] = (unsigned)(size_t)sQ;                  // lds_addr (bytes)
        g0[2] = (unsigned)gaddr;                       // global_addr[31:0]
        g0[3] = (unsigned)(gaddr >> 32) | (2u << 30);  // addr[56:32] | type=2
        i32x8 g1;
        g1[0] = (1 << 16)        // data_size = 2 bytes
              | (1 << 20)        // pad_enable
              | (7 << 22)        // pad_interval code 7 = 256 DWORDs (1024B)
              | (3 << 25);       // pad_amount  code 3 = 4 DWORDs (16B)
        g1[1] = 512 << 16;       // tensor_dim0 = 512 elements (bits 79:48)
        g1[2] = 32 << 16;        // tensor_dim1 = 32 rows       (bits 111:80)
        g1[3] = 512 << 16;       // tile_dim0   = 512           (bits 127:112)
        g1[4] = 32;              // tile_dim1   = 32            (bits 143:128)
        g1[5] = 4096;            // tensor_dim0_stride = 4096   (bits 207:160)
        g1[6] = 0;
        g1[7] = 0;
        i32x4 z4 = {0, 0, 0, 0};
        i32x8 z8 = {0, 0, 0, 0, 0, 0, 0, 0};
        __builtin_amdgcn_tensor_load_to_lds(g0, g1, z4, z4, z8, 0);
    }
    if (tid < FA_M) { sM[tid] = -INFINITY; sL[tid] = 0.0f; }
    __builtin_amdgcn_s_wait_tensorcnt(0);
    __syncthreads();

    const int smi = w >> 2, sni = w & 3;   // wave's S tile
    const int eW  = w * 64;                // wave's output column base
    const int half = lane >> 4, cl = lane & 15;

    f32x8 oacc[2][4] = {};

    for (int kv = 0; kv < qBase + FA_M; kv += FA_KV) {
        // ---- S tile: Q(32x512) x K^T chunk, one 16x16 tile per wave ----
        f32x8 s = {};
#pragma unroll
        for (int e = 0; e < 512; e += 32) {
            bf16x16 a  = frag_a_lds(sQ, SQ_ST, smi * 16, e, lane);
            bf16x16 bb = frag_b_glb(kB, 4096, kv + sni * 16, e, lane);
            s = __builtin_amdgcn_wmma_f32_16x16x32_bf16(
                false, a, false, bb, (short)0, s, false, false);
        }
        // causal mask + spill S to LDS
#pragma unroll
        for (int j = 0; j < 8; ++j) {
            int rl = smi * 16 + j + half * 8;       // local q row 0..31
            int kg = kv + sni * 16 + cl;            // global kv index
            float v = s[j];
            if (kg > qBase + rl) v = -INFINITY;
            sS[rl * SS_ST + sni * 16 + cl] = v;
        }
        __syncthreads();

        // ---- online softmax: 8 threads per row, shfl reductions ----
        {
            int r = tid >> 3, ts = tid & 7;
            const float* rowp = &sS[r * SS_ST + ts * 8];
            f32x4 v0 = *(const f32x4*)rowp;
            f32x4 v1 = *(const f32x4*)(rowp + 4);
            float x[8] = {v0[0], v0[1], v0[2], v0[3], v1[0], v1[1], v1[2], v1[3]};
            float mT = x[0];
#pragma unroll
            for (int i = 1; i < 8; ++i) mT = fmaxf(mT, x[i]);
            mT = fmaxf(mT, __shfl_xor(mT, 1, 8));
            mT = fmaxf(mT, __shfl_xor(mT, 2, 8));
            mT = fmaxf(mT, __shfl_xor(mT, 4, 8));
            float mOld = sM[r];
            float mNew = fmaxf(mOld, mT);
            float p[8], sum = 0.0f;
#pragma unroll
            for (int i = 0; i < 8; ++i) { p[i] = __expf(x[i] - mNew); sum += p[i]; }
            sum += __shfl_xor(sum, 1, 8);
            sum += __shfl_xor(sum, 2, 8);
            sum += __shfl_xor(sum, 4, 8);
            u32x4 pp;
#pragma unroll
            for (int i = 0; i < 4; ++i)
                pp[i] = (unsigned)f2bf(p[2*i]) | ((unsigned)f2bf(p[2*i+1]) << 16);
            *(u32x4*)(sP + r * SP_ST + ts * 8) = pp;
            if (ts == 0) {
                float corr = __expf(mOld - mNew);
                sF[r] = corr;
                sL[r] = sL[r] * corr + sum;
                sM[r] = mNew;
            }
        }
        __syncthreads();

        // ---- rescale O, accumulate P(32x64) @ V(64 x e-slice) ----
        float fr[2][8];
#pragma unroll
        for (int omi = 0; omi < 2; ++omi)
#pragma unroll
            for (int j = 0; j < 8; ++j)
                fr[omi][j] = sF[omi * 16 + j + half * 8];
#pragma unroll
        for (int omi = 0; omi < 2; ++omi)
#pragma unroll
            for (int oni = 0; oni < 4; ++oni)
#pragma unroll
                for (int j = 0; j < 8; ++j)
                    oacc[omi][oni][j] *= fr[omi][j];

#pragma unroll
        for (int ks = 0; ks < 2; ++ks) {
            bf16x16 a0 = frag_a_lds(sP, SP_ST, 0,  ks * 32, lane);
            bf16x16 a1 = frag_a_lds(sP, SP_ST, 16, ks * 32, lane);
#pragma unroll
            for (int oni = 0; oni < 4; ++oni) {
                bf16x16 bb = frag_b_glb(vB, 1024, eW + oni * 16, kv + ks * 32, lane);
                oacc[0][oni] = __builtin_amdgcn_wmma_f32_16x16x32_bf16(
                    false, a0, false, bb, (short)0, oacc[0][oni], false, false);
                oacc[1][oni] = __builtin_amdgcn_wmma_f32_16x16x32_bf16(
                    false, a1, false, bb, (short)0, oacc[1][oni], false, false);
            }
        }
        __syncthreads();
    }

    // ---- finalize: O /= l, store bf16 to [b][t][h][e] ----
    float inv[2][8];
#pragma unroll
    for (int omi = 0; omi < 2; ++omi)
#pragma unroll
        for (int j = 0; j < 8; ++j)
            inv[omi][j] = 1.0f / sL[omi * 16 + j + half * 8];
#pragma unroll
    for (int omi = 0; omi < 2; ++omi)
#pragma unroll
        for (int oni = 0; oni < 4; ++oni)
#pragma unroll
            for (int j = 0; j < 8; ++j) {
                int q = qBase + omi * 16 + j + half * 8;
                int e = eW + oni * 16 + cl;
                O[((size_t)(b * 1024 + q) * 8 + h) * 512 + e] =
                    f2bf(oacc[omi][oni][j] * inv[omi][j]);
            }
}

// ---------------------------------------------------------------------------
// Launch: 5 bf16 conversions -> 3 projection GEMMs -> flash -> unify GEMM.
// Workspace: Q(32M) | K(32M) | Vt(32M) | headOut(32M) | xb,wq,wk,wv,wu (4M ea)
// ---------------------------------------------------------------------------
extern "C" void kernel_launch(void* const* d_in, const int* in_sizes, int n_in,
                              void* d_out, int out_size, void* d_ws, size_t ws_size,
                              hipStream_t stream) {
    (void)in_sizes; (void)n_in; (void)out_size; (void)ws_size;
    const float* x   = (const float*)d_in[0];
    const float* w_k = (const float*)d_in[1];
    const float* w_q = (const float*)d_in[2];
    const float* w_v = (const float*)d_in[3];
    const float* w_u = (const float*)d_in[4];
    const float* b_u = (const float*)d_in[5];

    const size_t SEG = (size_t)16 * 1024 * 1024;   // elements
    const size_t WSEG = (size_t)2 * 1024 * 1024;   // elements (2M each)
    unsigned short* wsQ = (unsigned short*)d_ws;
    unsigned short* wsK = wsQ + SEG;
    unsigned short* wsV = wsK + SEG;
    unsigned short* wsA = wsV + SEG;
    unsigned short* xb  = wsA + SEG;
    unsigned short* wqb = xb  + WSEG;
    unsigned short* wkb = wqb + WSEG;
    unsigned short* wvb = wkb + WSEG;
    unsigned short* wub = wvb + WSEG;

    const float s = 0.21022410381342864f;          // 512^(-1/4)
    const int NCVT = 2 * 1024 * 1024;

    dim3 blk(256);
    cvt_bf16<<<1024, blk, 0, stream>>>(x,   xb,  NCVT);
    cvt_bf16<<<1024, blk, 0, stream>>>(w_q, wqb, NCVT);
    cvt_bf16<<<1024, blk, 0, stream>>>(w_k, wkb, NCVT);
    cvt_bf16<<<1024, blk, 0, stream>>>(w_v, wvb, NCVT);
    cvt_bf16<<<1024, blk, 0, stream>>>(w_u, wub, NCVT);

    dim3 gProj(32, 32);
    gemm_wmma<false, false><<<gProj, blk, 0, stream>>>(xb, wqb, wsQ, nullptr, s,    4096, 4096, 512);
    gemm_wmma<false, false><<<gProj, blk, 0, stream>>>(xb, wkb, wsK, nullptr, s,    4096, 4096, 512);
    gemm_wmma<false, true ><<<gProj, blk, 0, stream>>>(xb, wvb, wsV, nullptr, 1.0f, 4096, 4096, 512);

    dim3 gFA(32, 8, 4);
    flash_attn<<<gFA, blk, 0, stream>>>(wsQ, wsK, wsV, wsA);

    dim3 gU(32, 4);
    gemm_wmma<true, false><<<gU, blk, 0, stream>>>(wsA, wub, d_out, b_u, 1.0f, 4096, 512, 4096);
}
